// NeuralCDE_87969520157107
// MI455X (gfx1250) — matrix operand order repaired and checked by reading, not compiled, not run on hardware
//
#include <hip/hip_runtime.h>

typedef __attribute__((ext_vector_type(16))) _Float16 v16h;
typedef __attribute__((ext_vector_type(8)))  _Float16 v8h;
typedef __attribute__((ext_vector_type(8)))  float    v8f;

#define B_TOT 256
#define S_LEN 2048
#define I_DIM 24
#define H_DIM 64
#define O_DIM 10
#define NCOL  (I_DIM * H_DIM)   // 1536
#define NTILE (NCOL / 16)       // 96 N-tiles of 16 columns
#define WAVES 16
#define TPW   (NTILE / WAVES)   // 6 tiles per wave
#define BLK_B 16                // batch rows per block (== WMMA M)

__device__ __forceinline__ float fast_tanh(float x) {
    float y;
    asm volatile("v_tanh_f32 %0, %1" : "=v"(y) : "v"(x));
    return y;
}

__launch_bounds__(512, 1)
__global__ void ncde_kernel(const float* __restrict__ coeffs,   // [256,2048,24]
                            const float* __restrict__ W_init,   // [24,64]
                            const float* __restrict__ b_init,   // [64]
                            const float* __restrict__ W_func,   // [64,1536]
                            const float* __restrict__ b_func,   // [1536]
                            const float* __restrict__ W_out,    // [64,10]
                            const float* __restrict__ b_out,    // [10]
                            float* __restrict__ out)            // [256,10]
{
    __shared__ __align__(16) _Float16 fbuf[BLK_B][NCOL];   // 48 KB: tanh(z@Wf+bf)
    __shared__ __align__(16) float    zbase[BLK_B][H_DIM]; // z at step start
    __shared__ __align__(16) float    zacc[BLK_B][H_DIM];  // k1+2k2+2k3+k4
    __shared__ __align__(16) _Float16 zh[BLK_B][H_DIM];    // f16 copy of current z-input
    __shared__ __align__(16) float    dxs[BLK_B][I_DIM];   // dX for current interval
    __shared__ __align__(16) float    pred[BLK_B][16];     // final logits

    const int tid  = threadIdx.x;
    const int lane = tid & 31;
    const int wave = tid >> 5;
    const int b0   = blockIdx.x * BLK_B;

    const int nrow   = lane & 15;      // N within tile / M within tile
    const int hiHalf = lane >> 4;      // 0: lanes 0-15, 1: lanes 16-31

    // ------------------------------------------------------------------
    // Load this wave's W_func B-fragments (f16) into registers, once.
    // B frag k order chosen here (contiguous, matches A): half j of lane l = B[K][N],
    //   K = 16*(l>=16) + j (+32 per K-step), N = l%16.
    // ------------------------------------------------------------------
    v16h bfrag[TPW][2];
    float biasr[TPW];
#pragma unroll
    for (int t = 0; t < TPW; ++t) {
        const int ncolg = (wave * TPW + t) * 16 + nrow;
        biasr[t] = b_func[ncolg];
#pragma unroll
        for (int kk = 0; kk < 2; ++kk) {
            v16h f;
#pragma unroll
            for (int j = 0; j < 16; ++j) {
                const int kg = kk * 32 + 16 * hiHalf + j;
                f[j] = (_Float16)W_func[kg * NCOL + ncolg];
            }
            bfrag[t][kk] = f;
        }
    }

    // ------------------------------------------------------------------
    // z0 = coeffs[:,0,:] @ W_init + b_init.  Thread owns (b, h), (b, h+1).
    // ------------------------------------------------------------------
    const int myb = tid >> 5;          // 0..15
    const int myh = (tid & 31) * 2;    // 0,2,..,62
    {
        const float* cx = coeffs + (size_t)(b0 + myb) * S_LEN * I_DIM;
        float z0 = b_init[myh], z1 = b_init[myh + 1];
#pragma unroll
        for (int i = 0; i < I_DIM; ++i) {
            const float c = cx[i];
            z0 += c * W_init[i * H_DIM + myh];
            z1 += c * W_init[i * H_DIM + myh + 1];
        }
        zbase[myb][myh]     = z0;  zbase[myb][myh + 1] = z1;
        zh[myb][myh]        = (_Float16)z0;
        zh[myb][myh + 1]    = (_Float16)z1;
        zacc[myb][myh]      = 0.f; zacc[myb][myh + 1]  = 0.f;
    }

    // ------------------------------------------------------------------
    // dX register double-buffer: thread tid<384 owns (bb, ii).
    //   c_cur = coeffs[b, s+1, i]  (right knot of current interval)
    //   c_nxt = coeffs[b, s+2, i]  (prefetched one full step ahead)
    // ------------------------------------------------------------------
    const int  dxb = tid / I_DIM;      // valid for tid < 384
    const int  dxi = tid % I_DIM;
    const bool dxOwner = (tid < BLK_B * I_DIM);
    const float* cbase = coeffs + (size_t)(b0 + (dxOwner ? dxb : 0)) * S_LEN * I_DIM
                                + (dxOwner ? dxi : 0);
    float c_cur = 0.f, c_nxt = 0.f;
    if (dxOwner) {
        const float ca = cbase[0];              // s=0 knot
        c_cur = cbase[1 * I_DIM];               // s=1 knot
        dxs[dxb][dxi] = c_cur - ca;             // dX for interval 0
        c_nxt = cbase[2 * I_DIM];               // prefetch s=2 knot
    }
    __syncthreads();

    const float cin[4] = {0.5f, 0.5f, 1.0f, 0.0f};   // z-input scale for next k
    const float wk [4] = {1.0f, 2.0f, 2.0f, 1.0f};   // RK4 weights

    // ------------------------------------------------------------------
    // Scan over the 2047 intervals.
    // ------------------------------------------------------------------
#pragma unroll 1
    for (int s = 0; s < S_LEN - 1; ++s) {
        float dxr[I_DIM];
#pragma unroll
        for (int i = 0; i < I_DIM; ++i) dxr[i] = dxs[myb][i];

#pragma unroll
        for (int rk = 0; rk < 4; ++rk) {
            // ---- A fragments from zh (two 16B runs per K-step) ----
            const int r0 = hiHalf * 16;
            const v8h a0lo = *(const v8h*)&zh[nrow][r0];
            const v8h a0hi = *(const v8h*)&zh[nrow][r0 + 8];
            const v8h a1lo = *(const v8h*)&zh[nrow][r0 + 32];
            const v8h a1hi = *(const v8h*)&zh[nrow][r0 + 40];
            const v16h a0 = __builtin_shufflevector(a0lo, a0hi,
                0,1,2,3,4,5,6,7,8,9,10,11,12,13,14,15);
            const v16h a1 = __builtin_shufflevector(a1lo, a1hi,
                0,1,2,3,4,5,6,7,8,9,10,11,12,13,14,15);

            // ---- GEMM tiles + tanh + scatter to fbuf ----
#pragma unroll
            for (int t = 0; t < TPW; ++t) {
                v8f c;
#pragma unroll
                for (int r = 0; r < 8; ++r) c[r] = biasr[t];
                c = __builtin_amdgcn_wmma_f32_16x16x32_f16(
                        false, a0, false, bfrag[t][0], (short)0, c, false, false);
                c = __builtin_amdgcn_wmma_f32_16x16x32_f16(
                        false, a1, false, bfrag[t][1], (short)0, c, false, false);
                const int ncolg = (wave * TPW + t) * 16 + nrow;
#pragma unroll
                for (int r = 0; r < 8; ++r) {
                    const int m = r + 8 * hiHalf;          // D layout: M = r + 8*(l/16)
                    fbuf[m][ncolg] = (_Float16)fast_tanh(c[r]);
                }
            }
            __syncthreads();

            // ---- k[b,h] = sum_i f[b, h*24+i] * dx[b,i] ----
            const _Float16* f0 = &fbuf[myb][myh * I_DIM];
            const _Float16* f1 = &fbuf[myb][(myh + 1) * I_DIM];
            float k0 = 0.f, k1 = 0.f;
#pragma unroll
            for (int i = 0; i < I_DIM; ++i) {
                k0 += (float)f0[i] * dxr[i];
                k1 += (float)f1[i] * dxr[i];
            }

            const float za0 = zacc[myb][myh]     + wk[rk] * k0;
            const float za1 = zacc[myb][myh + 1] + wk[rk] * k1;
            zacc[myb][myh]     = za0;
            zacc[myb][myh + 1] = za1;

            if (rk < 3) {
                zh[myb][myh]     = (_Float16)(zbase[myb][myh]     + cin[rk] * k0);
                zh[myb][myh + 1] = (_Float16)(zbase[myb][myh + 1] + cin[rk] * k1);
            } else {
                const float zn0 = zbase[myb][myh]     + za0 * (1.0f / 6.0f);
                const float zn1 = zbase[myb][myh + 1] + za1 * (1.0f / 6.0f);
                zbase[myb][myh]     = zn0;  zbase[myb][myh + 1] = zn1;
                zh[myb][myh]        = (_Float16)zn0;
                zh[myb][myh + 1]    = (_Float16)zn1;
                zacc[myb][myh]      = 0.f;  zacc[myb][myh + 1]  = 0.f;

                // Publish dX for interval s+1 (prefetched a full step ago)
                // and kick off the load for interval s+2's right knot.
                if (dxOwner) {
                    dxs[dxb][dxi] = c_nxt - c_cur;
                    c_cur = c_nxt;
                    const int knot = (s + 3 < S_LEN) ? (s + 3) : (S_LEN - 1);
                    c_nxt = cbase[(size_t)knot * I_DIM];
                }
            }
            __syncthreads();
        }
    }

    // ------------------------------------------------------------------
    // pred = zT @ W_out + b_out ; softmax rows ; write out.
    // ------------------------------------------------------------------
    if (tid < BLK_B * O_DIM) {
        const int bb = tid / O_DIM, oo = tid % O_DIM;
        float acc = b_out[oo];
#pragma unroll
        for (int h = 0; h < H_DIM; ++h) acc += zbase[bb][h] * W_out[h * O_DIM + oo];
        pred[bb][oo] = acc;
    }
    __syncthreads();
    if (tid < BLK_B) {
        float m = pred[tid][0];
#pragma unroll
        for (int o = 1; o < O_DIM; ++o) m = fmaxf(m, pred[tid][o]);
        float sum = 0.f, e[O_DIM];
#pragma unroll
        for (int o = 0; o < O_DIM; ++o) { e[o] = __expf(pred[tid][o] - m); sum += e[o]; }
        const float inv = 1.0f / sum;
#pragma unroll
        for (int o = 0; o < O_DIM; ++o) out[(size_t)(b0 + tid) * O_DIM + o] = e[o] * inv;
    }
}

extern "C" void kernel_launch(void* const* d_in, const int* in_sizes, int n_in,
                              void* d_out, int out_size, void* d_ws, size_t ws_size,
                              hipStream_t stream) {
    (void)in_sizes; (void)n_in; (void)d_ws; (void)ws_size; (void)out_size;
    const float* coeffs = (const float*)d_in[0];
    const float* W_init = (const float*)d_in[1];
    const float* b_init = (const float*)d_in[2];
    const float* W_func = (const float*)d_in[3];
    const float* b_func = (const float*)d_in[4];
    const float* W_out  = (const float*)d_in[5];
    const float* b_out  = (const float*)d_in[6];
    float* out = (float*)d_out;

    ncde_kernel<<<dim3(B_TOT / BLK_B), dim3(WAVES * 32), 0, stream>>>(
        coeffs, W_init, b_init, W_func, b_func, W_out, b_out, out);
}
